// ViT_26216480375007
// MI455X (gfx1250) — matrix operand order repaired and checked by reading, not compiled, not run on hardware
//
#include <hip/hip_runtime.h>
#include <hip/hip_bf16.h>
#include <math.h>

// ---------------- CDNA5 WMMA types ----------------
typedef __attribute__((ext_vector_type(16))) __bf16 v16bf;
typedef __attribute__((ext_vector_type(8)))  float  v8f;

union Frag {
    v16bf v;
    uint4 q[2];
};

__device__ __forceinline__ v8f wmma_bf16(const Frag& a, const Frag& b, v8f c) {
    return __builtin_amdgcn_wmma_f32_16x16x32_bf16(false, a.v, false, b.v,
                                                   (short)0, c, false, false);
}

#define BATCH 32
#define NTOK  257          // 256 patches + cls
#define DIMC  768
#define MROWS (BATCH * NTOK)   // 8224
#define HEADS 12
#define DH    64
#define NKPAD 288          // keys padded to 9*32 for WMMA K-steps
#define NCHUNK (NKPAD / 32)    // 9 key chunks of 32

// =====================================================================
// Patch embedding: h[b, t, c]  (t=0 is cls token)
// =====================================================================
__global__ __launch_bounds__(256) void patch_embed_kernel(
    const float* __restrict__ x, const float* __restrict__ conv_w,
    const float* __restrict__ conv_b, const float* __restrict__ pos,
    const float* __restrict__ cls, float* __restrict__ h)
{
    int idx = blockIdx.x * 256 + threadIdx.x;
    if (idx >= BATCH * NTOK * DIMC) return;
    int c = idx % DIMC;
    int t = (idx / DIMC) % NTOK;
    int b = idx / (DIMC * NTOK);
    float v;
    if (t == 0) {
        v = cls[c] + pos[c];
    } else {
        const float* xp = x + (size_t)b * 4096 + (size_t)(t - 1) * 16;
        float acc = 0.f;
#pragma unroll
        for (int j = 0; j < 16; ++j) acc = fmaf(xp[j], conv_w[j * DIMC + c], acc);
        v = acc + conv_b[c] + pos[(size_t)t * DIMC + c];
    }
    h[idx] = v;
}

// =====================================================================
// LayerNorm over 768: one block per row; row_step lets us LN only cls rows
// =====================================================================
__global__ __launch_bounds__(256) void layernorm_kernel(
    const float* __restrict__ x, const float* __restrict__ w,
    const float* __restrict__ bv, __bf16* __restrict__ out, int row_step)
{
    __shared__ float sh1[256], sh2[256];
    int tid = threadIdx.x;
    const float* xr = x + (size_t)blockIdx.x * row_step * DIMC;
    float vals[3], s1 = 0.f, s2 = 0.f;
#pragma unroll
    for (int i = 0; i < 3; ++i) {
        float v = xr[tid + i * 256];
        vals[i] = v; s1 += v; s2 += v * v;
    }
    sh1[tid] = s1; sh2[tid] = s2;
    __syncthreads();
    for (int off = 128; off > 0; off >>= 1) {
        if (tid < off) { sh1[tid] += sh1[tid + off]; sh2[tid] += sh2[tid + off]; }
        __syncthreads();
    }
    float mu = sh1[0] * (1.f / DIMC);
    float var = sh2[0] * (1.f / DIMC) - mu * mu;
    float rstd = rsqrtf(var + 1e-5f);
    __bf16* orow = out + (size_t)blockIdx.x * DIMC;
#pragma unroll
    for (int i = 0; i < 3; ++i) {
        int c = tid + i * 256;
        orow[c] = (__bf16)((vals[i] - mu) * rstd * w[c] + bv[c]);
    }
}

// =====================================================================
// WMMA GEMM: C[M,N] = A[M,K](bf16) * W[K,N](f32, converted to bf16 in LDS)
// Block tile 64x128, BK=32, 8 waves (4x2), each wave -> 16x64 of C.
// =====================================================================
template <bool HAS_BIAS, bool GELU, bool RESID, bool OUT_BF16>
__global__ __launch_bounds__(256) void gemm_wmma_kernel(
    const __bf16* __restrict__ A, const float* __restrict__ W,
    const float* __restrict__ bias, const float* __restrict__ resid,
    float* __restrict__ outf, __bf16* __restrict__ outb,
    int M, int N, int K)
{
    __shared__ __bf16 As[64][32];     // A tile, row-major
    __shared__ __bf16 Ws[128][32];    // W tile transposed: [n][k]

    int tid  = threadIdx.x;
    int lane = tid & 31;
    int wid  = tid >> 5;
    int wm   = wid >> 1;              // 0..3
    int wn   = wid & 1;               // 0..1
    int bm   = blockIdx.y * 64;
    int bn   = blockIdx.x * 128;

    v8f vzero = {0.f,0.f,0.f,0.f,0.f,0.f,0.f,0.f};
    v8f acc[4];
#pragma unroll
    for (int j = 0; j < 4; ++j) acc[j] = vzero;

    for (int k0 = 0; k0 < K; k0 += 32) {
        // --- stage A: 64x32 bf16, 16B per thread ---
        {
            int idx = tid * 8;
            int r = idx >> 5, c = idx & 31;
            int gr = bm + r;
            uint4 val = make_uint4(0u, 0u, 0u, 0u);
            if (gr < M) val = *(const uint4*)(A + (size_t)gr * K + k0 + c);
            *(uint4*)(&As[r][c]) = val;
        }
        // --- stage W transposed + f32->bf16 convert ---
#pragma unroll 4
        for (int i = 0; i < 16; ++i) {
            int idx = tid + i * 256;      // 0..4095
            int kk = idx >> 7;            // 0..31
            int n  = idx & 127;
            int gn = bn + n;
            float w = 0.f;
            if (gn < N) w = W[(size_t)(k0 + kk) * N + gn];
            Ws[n][kk] = (__bf16)w;
        }
        if (k0 + 32 < K) {   // prefetch next A tile (global_prefetch_b8)
            int idx = tid * 8;
            int gr = bm + (idx >> 5);
            if (gr < M) __builtin_prefetch(A + (size_t)gr * K + k0 + 32 + (idx & 31), 0, 3);
        }
        __syncthreads();

        // --- A fragment: two contiguous 8-elem K runs per lane ---
        Frag a;
        {
            int r  = wm * 16 + (lane & 15);
            int kb = (lane & 16) ? 16 : 0;
            a.q[0] = *(const uint4*)(&As[r][kb]);
            a.q[1] = *(const uint4*)(&As[r][kb + 8]);
        }
        int kb2 = (lane & 16) ? 16 : 0;
#pragma unroll
        for (int j = 0; j < 4; ++j) {
            Frag b;
            int n = wn * 64 + j * 16 + (lane & 15);
            b.q[0] = *(const uint4*)(&Ws[n][kb2]);
            b.q[1] = *(const uint4*)(&Ws[n][kb2 + 8]);
            acc[j] = wmma_bf16(a, b, acc[j]);
        }
        __syncthreads();
    }

    // --- epilogue: C layout -> global ---
    int lm = (lane & 16) ? 8 : 0;
    int lc = lane & 15;
#pragma unroll
    for (int j = 0; j < 4; ++j) {
        int gc = bn + wn * 64 + j * 16 + lc;
        if (gc >= N) continue;
        float bvv = 0.f;
        if (HAS_BIAS) bvv = bias[gc];
#pragma unroll
        for (int r = 0; r < 8; ++r) {
            int gr = bm + wm * 16 + lm + r;
            if (gr >= M) continue;
            float v = acc[j][r] + bvv;
            if (GELU)  v = 0.5f * v * (1.0f + erff(v * 0.7071067811865476f));
            if (RESID) v += resid[(size_t)gr * N + gc];
            if (OUT_BF16) outb[(size_t)gr * N + gc] = (__bf16)v;
            else          outf[(size_t)gr * N + gc] = v;
        }
    }
}

// =====================================================================
// Fused flash attention: one block per (batch, head).
// K staged global->LDS with GLOBAL_LOAD_ASYNC_TO_LDS_B128 (ASYNCcnt),
// V^T staged via 16B loads + transposed ds stores. Online softmax over
// 9 chunks of 32 keys keeps live state at ~150 VGPRs (no spills).
// qkv: bf16 [B*257, 2304] ; obuf: bf16 [B*257, 768]
// =====================================================================
__global__ __launch_bounds__(256) void attention_kernel(
    const __bf16* __restrict__ qkv, __bf16* __restrict__ obuf)
{
    extern __shared__ char smem_raw[];
    __bf16* Ks = (__bf16*)smem_raw;            // [288][64] row-major (key,d)
    __bf16* Vt = Ks + NKPAD * DH;              // [64][288]  (d,key)
    __bf16* Ps = Vt + NKPAD * DH;              // 8 waves x [16][32] P chunk

    int bh = blockIdx.x;
    int b = bh / HEADS, h = bh % HEADS;
    int tid = threadIdx.x, lane = tid & 31, wid = tid >> 5;
    const __bf16* base = qkv + (size_t)b * NTOK * 2304;

    // ---- K rows 0..256: async global->LDS 16B copies (257*8 = 2056 chunks).
    // LDS offset = low 32 bits of the generic shared pointer (LDS aperture
    // truncation, ISA 10.2); each wave waits on its own ASYNCcnt below.
    unsigned ks_base = (unsigned)(uintptr_t)(void*)Ks;
    for (int chunk = tid; chunk < NTOK * 8; chunk += 256) {
        int tok = chunk >> 3, part = chunk & 7;
        const __bf16* gp = base + (size_t)tok * 2304 + 768 + h * DH + part * 8;
        unsigned lds = ks_base + (unsigned)(tok * DH + part * 8) * 2u;
        asm volatile("global_load_async_to_lds_b128 %0, %1, off"
                     :: "v"(lds), "v"(gp) : "memory");
    }
    // zero-pad K rows 257..287
    for (int i = tid; i < (NKPAD - NTOK) * DH; i += 256)
        Ks[NTOK * DH + i] = (__bf16)0.f;
    // ---- V transposed: one 16B global load per thread-chunk, 8 ds stores
    for (int chunk = tid; chunk < NKPAD * 8; chunk += 256) {
        int tok = chunk >> 3, part = chunk & 7;
        union { uint4 q; __bf16 e[8]; } u;
        u.q = make_uint4(0u, 0u, 0u, 0u);
        if (tok < NTOK)
            u.q = *(const uint4*)(base + (size_t)tok * 2304 + 1536 + h * DH + part * 8);
#pragma unroll
        for (int i = 0; i < 8; ++i)
            Vt[(part * 8 + i) * NKPAD + tok] = u.e[i];
    }
    asm volatile("s_wait_asynccnt 0" ::: "memory");
    __syncthreads();

    v8f vzero = {0.f,0.f,0.f,0.f,0.f,0.f,0.f,0.f};
    int kb  = (lane & 16) ? 16 : 0;   // A-frag K-run base
    int kb2 = (lane & 16) ? 16 : 0;   // B-frag K-run base
    int lc  = lane & 15;
    int lm  = (lane & 16) ? 8 : 0;
    __bf16* pw = Ps + wid * 16 * 32;  // this wave's P staging tile

    // 24 m-tiles total -> uniform 3 iterations per wave
    for (int it = 0; it < 3; ++it) {
        int mt = wid + it * 8;        // 0..23; tiles >=17 fully masked on store

        // Q fragments (d 0..31 and 32..63) straight from global, zero for pad rows
        Frag qf[2];
        {
            int row = mt * 16 + lc;
            if (row < NTOK) {
                const __bf16* qp = base + (size_t)row * 2304 + h * DH;
#pragma unroll
                for (int s = 0; s < 2; ++s) {
                    qf[s].q[0] = *(const uint4*)(qp + s * 32 + kb);
                    qf[s].q[1] = *(const uint4*)(qp + s * 32 + kb + 8);
                }
            } else {
                uint4 z = make_uint4(0u,0u,0u,0u);
                qf[0].q[0]=z; qf[0].q[1]=z; qf[1].q[0]=z; qf[1].q[1]=z;
            }
        }

        v8f o[4];
#pragma unroll
        for (int j = 0; j < 4; ++j) o[j] = vzero;
        float rmax[8], rsum[8];
#pragma unroll
        for (int r = 0; r < 8; ++r) { rmax[r] = -__builtin_inff(); rsum[r] = 0.f; }

        for (int kt = 0; kt < NCHUNK; ++kt) {
            // ---- S chunk = Q K^T for 32 keys (2 column tiles x 2 K-steps) ----
            v8f s[2];
            s[0] = vzero; s[1] = vzero;
#pragma unroll
            for (int j = 0; j < 2; ++j) {
                int key = kt * 32 + j * 16 + lc;
#pragma unroll
                for (int ks = 0; ks < 2; ++ks) {
                    Frag kf;
                    const __bf16* kp = Ks + (size_t)key * DH + ks * 32 + kb2;
                    kf.q[0] = *(const uint4*)(kp);
                    kf.q[1] = *(const uint4*)(kp + 8);
                    s[j] = wmma_bf16(qf[ks], kf, s[j]);
                }
            }

            // ---- scale + mask + chunk row-max ----
            float cmax[8];
#pragma unroll
            for (int r = 0; r < 8; ++r) cmax[r] = -__builtin_inff();
#pragma unroll
            for (int j = 0; j < 2; ++j) {
                bool ok = (kt * 32 + j * 16 + lc) < NTOK;
#pragma unroll
                for (int r = 0; r < 8; ++r) {
                    float v = s[j][r] * 0.125f;        // dh^-0.5
                    if (!ok) v = -__builtin_inff();
                    s[j][r] = v;
                    cmax[r] = fmaxf(cmax[r], v);
                }
            }
#pragma unroll
            for (int r = 0; r < 8; ++r)
#pragma unroll
                for (int m = 1; m < 16; m <<= 1)
                    cmax[r] = fmaxf(cmax[r], __shfl_xor(cmax[r], m, 32));

            // ---- online rescale ----
            float factor[8];
#pragma unroll
            for (int r = 0; r < 8; ++r) {
                float nm = fmaxf(rmax[r], cmax[r]);
                factor[r] = __expf(rmax[r] - nm);      // exp(-inf)=0 on first chunk
                rmax[r] = nm;
            }
            float csum[8];
#pragma unroll
            for (int r = 0; r < 8; ++r) csum[r] = 0.f;
#pragma unroll
            for (int j = 0; j < 2; ++j)
#pragma unroll
                for (int r = 0; r < 8; ++r) {
                    float e = __expf(s[j][r] - rmax[r]);
                    s[j][r] = e;
                    csum[r] += e;
                }
#pragma unroll
            for (int r = 0; r < 8; ++r) {
#pragma unroll
                for (int m = 1; m < 16; m <<= 1)
                    csum[r] += __shfl_xor(csum[r], m, 32);
                rsum[r] = rsum[r] * factor[r] + csum[r];
            }
#pragma unroll
            for (int j = 0; j < 4; ++j)
#pragma unroll
                for (int r = 0; r < 8; ++r)
                    o[j][r] *= factor[r];

            // ---- P chunk: C-layout regs -> wave-private LDS [16][32] ----
#pragma unroll
            for (int j = 0; j < 2; ++j)
#pragma unroll
                for (int r = 0; r < 8; ++r)
                    pw[(lm + r) * 32 + j * 16 + lc] = (__bf16)s[j][r];
            __builtin_amdgcn_wave_barrier();   // LDS is in-order per wave; pin scheduler

            // ---- O += P (16x32) @ V (32x64): one K-step, 4 dh tiles ----
            Frag pa;
            {
                const __bf16* pp = pw + lc * 32 + kb;
                pa.q[0] = *(const uint4*)(pp);
                pa.q[1] = *(const uint4*)(pp + 8);
            }
#pragma unroll
            for (int j = 0; j < 4; ++j) {
                Frag vb;
                int d = j * 16 + lc;
                const __bf16* vp = Vt + (size_t)d * NKPAD + kt * 32 + kb2;
                vb.q[0] = *(const uint4*)(vp);
                vb.q[1] = *(const uint4*)(vp + 8);
                o[j] = wmma_bf16(pa, vb, o[j]);
            }
            __builtin_amdgcn_wave_barrier();   // keep next chunk's P stores behind reads
        }

        // ---- normalize + store O tile (bf16) for the out-projection GEMM ----
        float rinv[8];
#pragma unroll
        for (int r = 0; r < 8; ++r) rinv[r] = 1.0f / rsum[r];
#pragma unroll
        for (int j = 0; j < 4; ++j) {
            int d = h * DH + j * 16 + lc;
#pragma unroll
            for (int r = 0; r < 8; ++r) {
                int tok = mt * 16 + lm + r;
                if (tok < NTOK)
                    obuf[((size_t)b * NTOK + tok) * DIMC + d] = (__bf16)(o[j][r] * rinv[r]);
            }
        }
    }
}

// =====================================================================
// Host-side launch
// =====================================================================
extern "C" void kernel_launch(void* const* d_in, const int* in_sizes, int n_in,
                              void* d_out, int out_size, void* d_ws, size_t ws_size,
                              hipStream_t stream)
{
    (void)in_sizes; (void)n_in; (void)out_size; (void)ws_size;

    const float* x        = (const float*)d_in[0];
    const float* conv_w   = (const float*)d_in[1];
    const float* conv_b   = (const float*)d_in[2];
    const float* pos      = (const float*)d_in[3];
    const float* cls_tok  = (const float*)d_in[4];
    const float* ln1_w    = (const float*)d_in[5];
    const float* ln1_b    = (const float*)d_in[6];
    const float* qkv_w    = (const float*)d_in[7];
    const float* out_w    = (const float*)d_in[8];
    const float* out_b    = (const float*)d_in[9];
    const float* ln2_w    = (const float*)d_in[10];
    const float* ln2_b    = (const float*)d_in[11];
    const float* mlp_w1   = (const float*)d_in[12];
    const float* mlp_b1   = (const float*)d_in[13];
    const float* mlp_w2   = (const float*)d_in[14];
    const float* mlp_b2   = (const float*)d_in[15];
    const float* cls_ln_w = (const float*)d_in[16];
    const float* cls_ln_b = (const float*)d_in[17];
    const float* head_w   = (const float*)d_in[18];
    const float* head_b   = (const float*)d_in[19];
    float* outp = (float*)d_out;

    // workspace carve-up
    float*  hbuf = (float*)d_ws;                               // [8224,768] f32
    __bf16* zbuf = (__bf16*)(hbuf + (size_t)MROWS * DIMC);     // [8224,768]
    __bf16* qkvb = zbuf + (size_t)MROWS * DIMC;                // [8224,2304]
    __bf16* obuf = qkvb + (size_t)MROWS * 2304;                // [8224,768]
    __bf16* mbuf = obuf + (size_t)MROWS * DIMC;                // [8224,3072]
    __bf16* clsz = mbuf + (size_t)MROWS * 3072;                // [32,768]

    const int attn_shm = (2 * NKPAD * DH + 8 * 16 * 32) * (int)sizeof(__bf16); // 81920
    hipFuncSetAttribute((const void*)attention_kernel,
                        hipFuncAttributeMaxDynamicSharedMemorySize, attn_shm);

    // patch embedding
    {
        int total = BATCH * NTOK * DIMC;
        patch_embed_kernel<<<(total + 255) / 256, 256, 0, stream>>>(
            x, conv_w, conv_b, pos, cls_tok, hbuf);
    }

    const int gm = (MROWS + 63) / 64;   // 129
    for (int l = 0; l < 12; ++l) {
        // LN1 -> z
        layernorm_kernel<<<MROWS, 256, 0, stream>>>(
            hbuf, ln1_w + (size_t)l * DIMC, ln1_b + (size_t)l * DIMC, zbuf, 1);
        // qkv = z @ Wqkv (no bias) -> bf16
        gemm_wmma_kernel<false, false, false, true>
            <<<dim3(2304 / 128, gm), 256, 0, stream>>>(
                zbuf, qkv_w + (size_t)l * DIMC * 2304, nullptr, nullptr,
                nullptr, qkvb, MROWS, 2304, DIMC);
        // fused attention
        attention_kernel<<<BATCH * HEADS, 256, attn_shm, stream>>>(qkvb, obuf);
        // h = o @ Wout + bout + h
        gemm_wmma_kernel<true, false, true, false>
            <<<dim3(DIMC / 128, gm), 256, 0, stream>>>(
                obuf, out_w + (size_t)l * DIMC * DIMC, out_b + (size_t)l * DIMC,
                hbuf, hbuf, nullptr, MROWS, DIMC, DIMC);
        // LN2 -> z
        layernorm_kernel<<<MROWS, 256, 0, stream>>>(
            hbuf, ln2_w + (size_t)l * DIMC, ln2_b + (size_t)l * DIMC, zbuf, 1);
        // m = gelu(z @ W1 + b1) -> bf16
        gemm_wmma_kernel<true, true, false, true>
            <<<dim3(3072 / 128, gm), 256, 0, stream>>>(
                zbuf, mlp_w1 + (size_t)l * DIMC * 3072, mlp_b1 + (size_t)l * 3072,
                nullptr, nullptr, mbuf, MROWS, 3072, DIMC);
        // h = m @ W2 + b2 + h
        gemm_wmma_kernel<true, false, true, false>
            <<<dim3(DIMC / 128, gm), 256, 0, stream>>>(
                mbuf, mlp_w2 + (size_t)l * 3072 * DIMC, mlp_b2 + (size_t)l * DIMC,
                hbuf, hbuf, nullptr, MROWS, DIMC, 3072);
    }

    // final LN over cls rows (row b*257), then head GEMM
    layernorm_kernel<<<BATCH, 256, 0, stream>>>(hbuf, cls_ln_w, cls_ln_b, clsz, NTOK);
    gemm_wmma_kernel<true, false, false, false>
        <<<dim3((1000 + 127) / 128, 1), 256, 0, stream>>>(
            clsz, head_w, head_b, nullptr, outp, nullptr, BATCH, 1000, DIMC);
}
